// ComprehensiveWindowAwareLoss_52733608460401
// MI455X (gfx1250) — matrix operand order repaired and checked by reading, not compiled, not run on hardware
//
#include <hip/hip_runtime.h>
#include <hip/hip_bf16.h>
#include <math.h>

// ---------------- problem constants ----------------
#define BATCH 16
#define CHN   3
#define HH    512
#define WW    512
#define LP    7            // left pad of filtered dimension
#define PW    544          // padded length: 7 + 512 + 25 (multiple of 16)
#define NTILE (BATCH*32*32)   // 16384 16x16 tiles
#define NBLK  (NTILE/8)       // 2048 blocks, 8 waves each
#define EPSV  1e-8f

typedef __attribute__((ext_vector_type(16))) _Float16 v16h;
typedef __attribute__((ext_vector_type(8)))  float    v8f;

// ---------------- helpers ----------------
__device__ __forceinline__ float sigm(float x) { return 1.0f / (1.0f + __expf(-x)); }

// Band matrix A (16x32, f16): A[m][k] = 1 if m <= k <= m+14 else 0.
// Contiguous-k layout used here (same as B) for A 16x32: lane L -> M = L%16, half = L/16;
// packed element e (0..15): k = half*16+e.
__device__ __forceinline__ v16h make_band(int lane) {
  int m = lane & 15, half = lane >> 4;
  v16h a;
#pragma unroll
  for (int e = 0; e < 16; ++e) {
    int k = half * 16 + e;
    a[e] = (k >= m && k <= m + 14) ? (_Float16)1.0f : (_Float16)0.0f;
  }
  return a;
}

// ---------------- kernel 0: init flag ----------------
__global__ void k_flag_init(int* flag) {
  if (blockIdx.x == 0 && threadIdx.x == 0) *flag = 0;
}

// ---------------- kernel 1: any(source < 0) ----------------
__global__ void k_any_negative(const float* __restrict__ src, int n, int* flag) {
  __shared__ int s;
  if (threadIdx.x == 0) s = 0;
  __syncthreads();
  int any = 0;
  for (int i = blockIdx.x * blockDim.x + threadIdx.x; i < n; i += gridDim.x * blockDim.x)
    any |= (src[i] < 0.0f) ? 1 : 0;
  if (any) atomicOr(&s, 1);
  __syncthreads();
  if (threadIdx.x == 0 && s) atomicOr(flag, 1);   // idempotent -> deterministic
}

// ---------------- kernel 2: per-pixel mask -> padded f16 [B][H][PW] ----------------
__global__ void k_mask(const float* __restrict__ src, const int* __restrict__ flag,
                       _Float16* __restrict__ mask) {
  long idx = (long)blockIdx.x * blockDim.x + threadIdx.x;
  const long total = (long)BATCH * HH * PW;
  if (idx >= total) return;
  int x = (int)(idx % PW);
  long t = idx / PW;
  int r = (int)(t % HH);
  int b = (int)(t / HH);
  float m = 0.0f;
  int c = x - LP;
  if (c >= 0 && c < WW) {
    const long plane = (long)HH * WW;
    long base = (long)b * CHN * plane + (long)r * WW + c;
    float s0 = src[base], s1 = src[base + plane], s2 = src[base + 2 * plane];
    if (*flag) { s0 = (s0 + 1.0f) * 0.5f; s1 = (s1 + 1.0f) * 0.5f; s2 = (s2 + 1.0f) * 0.5f; }
    float br  = 0.299f * s0 + 0.587f * s1 + 0.114f * s2;
    float bm  = sigm(20.0f * (br - 0.65f));
    float mx  = fmaxf(s0, fmaxf(s1, s2));
    float mn  = fminf(s0, fminf(s1, s2));
    float lsm = sigm(20.0f * (0.15f - (mx - mn)));
    m = bm * lsm;
  }
  mask[idx] = (_Float16)m;
}

// ---------------- kernel 3: zero the pad columns of hfiltT [B][W][PW] ----------------
__global__ void k_zero_pads(_Float16* __restrict__ hf) {
  const int PADN = PW - HH;  // 32
  int idx = blockIdx.x * blockDim.x + threadIdx.x;
  if (idx >= BATCH * WW * PADN) return;
  int p = idx % PADN;
  int t = idx / PADN;               // b*W + c
  int x = (p < LP) ? p : (HH + p);  // [0,7) and [519,544)
  hf[(long)t * PW + x] = (_Float16)0.0f;
}

// ---------------- kernel 4: horizontal box pass via WMMA, store transposed ----------------
// in : mask  [B][H][PW]  (padded along columns)
// out: hfiltT[B][W][PW]  (padded along rows; value = sum of 15 taps, transposed)
__global__ void k_hpass(const _Float16* __restrict__ mask, _Float16* __restrict__ hf) {
  int lane = threadIdx.x & 31, wave = threadIdx.x >> 5;
  int tile = blockIdx.x * 8 + wave;             // 0..16383
  int b = tile >> 10, rem = tile & 1023;
  int r0 = (rem >> 5) * 16, c0 = (rem & 31) * 16;
  int half = lane >> 4, ln = lane & 15;

  // B(32x16): lane -> N = row offset ln, K = 16*half + e  (16 contiguous f16, 32B aligned)
  const _Float16* bp = mask + ((long)b * HH + r0 + ln) * PW + (c0 + 16 * half);
  v16h bmat = *(const v16h*)bp;
  v16h amat = make_band(lane);
  v8f  cacc = {};
  v8f  d = __builtin_amdgcn_wmma_f32_16x16x32_f16(false, amat, false, bmat,
                                                  (short)0, cacc, false, false);
  // D: M (=output col offset) = vr + 8*half, N (=row offset) = ln. Store transposed.
#pragma unroll
  for (int vr = 0; vr < 8; ++vr) {
    int cc = c0 + half * 8 + vr;
    hf[((long)b * WW + cc) * PW + LP + r0 + ln] = (_Float16)d[vr];
  }
}

// ---------------- kernel 5: vertical box pass via WMMA + fused loss ----------------
__global__ void k_vpass_loss(const _Float16* __restrict__ hf,
                             const float* __restrict__ pred,
                             const float* __restrict__ targ,
                             const float* __restrict__ src,
                             float* __restrict__ partials) {
  __shared__ float red[8][3];
  int lane = threadIdx.x & 31, wave = threadIdx.x >> 5;
  int tile = blockIdx.x * 8 + wave;
  int b = tile >> 10, rem = tile & 1023;
  int r0 = (rem >> 5) * 16, ct0 = (rem & 31) * 16;
  int half = lane >> 4, ln = lane & 15;

  const _Float16* bp = hf + ((long)b * WW + ct0 + ln) * PW + (r0 + 16 * half);
  v16h bmat = *(const v16h*)bp;
  v16h amat = make_band(lane);
  v8f  cacc = {};
  v8f  d = __builtin_amdgcn_wmma_f32_16x16x32_f16(false, amat, false, bmat,
                                                  (short)0, cacc, false, false);

  const long plane = (long)HH * WW;
  float S1 = 0.0f, S2 = 0.0f, S3 = 0.0f;
#pragma unroll
  for (int vr = 0; vr < 8; ++vr) {
    float m = d[vr] * (1.0f / 225.0f);          // finished window_mask value
    int r = r0 + half * 8 + vr;
    int cc = ct0 + ln;
    long base = (long)b * CHN * plane + (long)r * WW + cc;
    float p0 = pred[base], p1 = pred[base + plane], p2 = pred[base + 2 * plane];
    float t0 = targ[base], t1 = targ[base + plane], t2 = targ[base + 2 * plane];
    float s0 = src[base],  s1 = src[base + plane],  s2 = src[base + 2 * plane];

    float a = fabsf(p0 - t0) + fabsf(p1 - t1) + fabsf(p2 - t2);
    S1 += a;
    S2 += a * (1.0f + 4.0f * m);               // weight = BASE + (WIN-BASE)*mask

    float st0 = t0 - s0, st1 = t1 - s1, st2 = t2 - s2;
    float sp0 = p0 - s0, sp1 = p1 - s1, sp2 = p2 - s2;
    float stm = sqrtf(st0 * st0 + st1 * st1 + st2 * st2);
    float spm = sqrtf(sp0 * sp0 + sp1 * sp1 + sp2 * sp2);
    float ist = 1.0f / fmaxf(stm, EPSV);
    float isp = 1.0f / fmaxf(spm, EPSV);
    float align = (st0 * sp0 + st1 * sp1 + st2 * sp2) * ist * isp;
    float magl  = fabsf(spm / (stm + EPSV) - 1.0f);
    S3 += m * (1.0f - align + 0.5f * magl);
  }

  // wave32 reduction
  for (int off = 16; off > 0; off >>= 1) {
    S1 += __shfl_down(S1, off, 32);
    S2 += __shfl_down(S2, off, 32);
    S3 += __shfl_down(S3, off, 32);
  }
  if (lane == 0) { red[wave][0] = S1; red[wave][1] = S2; red[wave][2] = S3; }
  __syncthreads();
  if (threadIdx.x == 0) {
    float a = 0.0f, bb = 0.0f, c2 = 0.0f;
    for (int w = 0; w < 8; ++w) { a += red[w][0]; bb += red[w][1]; c2 += red[w][2]; }
    partials[blockIdx.x * 3 + 0] = a;
    partials[blockIdx.x * 3 + 1] = bb;
    partials[blockIdx.x * 3 + 2] = c2;
  }
}

// ---------------- kernel 6: deterministic final reduction ----------------
__global__ void k_final(const float* __restrict__ partials, float* __restrict__ out) {
  __shared__ float sh[256];
  float acc[3] = {0.0f, 0.0f, 0.0f};
  for (int j = threadIdx.x; j < NBLK; j += 256) {
    acc[0] += partials[j * 3 + 0];
    acc[1] += partials[j * 3 + 1];
    acc[2] += partials[j * 3 + 2];
  }
  float res[3];
  for (int s = 0; s < 3; ++s) {
    sh[threadIdx.x] = acc[s];
    __syncthreads();
    for (int off = 128; off > 0; off >>= 1) {
      if ((int)threadIdx.x < off) sh[threadIdx.x] += sh[threadIdx.x + off];
      __syncthreads();
    }
    res[s] = sh[0];
    __syncthreads();
  }
  if (threadIdx.x == 0) {
    const float N3 = (float)BATCH * CHN * HH * WW;
    const float N1 = (float)BATCH * HH * WW;
    // total = 1*l1 + 3*win_l1 + 1*(2*mean_color)
    out[0] = res[0] / N3 + 3.0f * (res[1] / N3) + 2.0f * (res[2] / N1);
  }
}

// ---------------- launcher ----------------
extern "C" void kernel_launch(void* const* d_in, const int* in_sizes, int n_in,
                              void* d_out, int out_size, void* d_ws, size_t ws_size,
                              hipStream_t stream) {
  (void)in_sizes; (void)n_in; (void)out_size; (void)ws_size;
  const float* pred = (const float*)d_in[0];
  const float* targ = (const float*)d_in[1];
  const float* src  = (const float*)d_in[2];
  float* out = (float*)d_out;

  char* ws = (char*)d_ws;
  int*   flag     = (int*)ws;                                  // 4 B
  float* partials = (float*)(ws + 256);                        // 2048*3*4 = 24 KB
  const size_t PADSZ = (size_t)BATCH * HH * PW * sizeof(_Float16);  // 8.9 MB each
  _Float16* mask = (_Float16*)(ws + 32768);
  _Float16* hf   = (_Float16*)(ws + 32768 + PADSZ);

  const int NPIX_PAD = BATCH * HH * PW;      // 4,456,448
  const int NPAD_HF  = BATCH * WW * (PW - HH);

  k_flag_init   <<<1, 1, 0, stream>>>(flag);
  k_any_negative<<<1024, 256, 0, stream>>>(src, BATCH * CHN * HH * WW, flag);
  k_mask        <<<(NPIX_PAD + 255) / 256, 256, 0, stream>>>(src, flag, mask);
  k_zero_pads   <<<(NPAD_HF + 255) / 256, 256, 0, stream>>>(hf);
  k_hpass       <<<NBLK, 256, 0, stream>>>(mask, hf);
  k_vpass_loss  <<<NBLK, 256, 0, stream>>>(hf, pred, targ, src, partials);
  k_final       <<<1, 256, 0, stream>>>(partials, out);
}
